// SkipLSTM_32933809226276
// MI455X (gfx1250) — matrix operand order repaired and checked by reading, not compiled, not run on hardware
//
#include <hip/hip_runtime.h>
#include <hip/hip_bf16.h>
#include <math.h>

// SkipLSTM for MI455X (gfx1250, wave32, WMMA bf16 path).
// B=128, S=512, IN=HS=1024. Weights converted once to bf16 W^T so WMMA
// B-fragments are contiguous per lane; they stay L2-resident (32 MB << 192 MB
// L2) across all 512 sequential steps. The recurrence is latency-bound, so
// the cell kernel stages whole 32x512 activation tiles into LDS per phase and
// then runs branch-free, barrier-free WMMA iterations.

#define HS    1024
#define SEQ   512
#define BATCH 128

#define LDS_STRIDE 520   // 32 rows x 520 bf16 = 33280 B; 520*2B = 1040 B
                         // => bank step 4 per row, 16 readers conflict-free

typedef __attribute__((ext_vector_type(16))) __bf16 v16bf;
typedef __attribute__((ext_vector_type(8)))  __bf16 v8bf;
typedef __attribute__((ext_vector_type(8)))  float  v8f;

__device__ __forceinline__ float sigmoidf_(float x) {
    return 1.0f / (1.0f + __expf(-x));
}

// ---------------------------------------------------------------------------
// One-time: fp32 (K=1024 rows, N=4096 cols, row-major) -> bf16 W^T [n][k]
// ---------------------------------------------------------------------------
__global__ __launch_bounds__(256) void wt_transpose_bf16(
    const float* __restrict__ in, __bf16* __restrict__ out)
{
    const int idx = blockIdx.x * 256 + threadIdx.x;   // 0 .. 4096*1024-1
    const int n = idx >> 10;
    const int k = idx & 1023;
    out[idx] = (__bf16)in[(long)k * 4096 + n];
}

__global__ void u_init(float* u_t, float* u_bin)
{
    const int i = threadIdx.x;   // 128 threads
    u_t[i]   = 1.0f;
    u_bin[i] = 1.0f;
}

// ---------------------------------------------------------------------------
// Fused LSTM cell: A = act@U + h@V + b (bf16 WMMA, fp32 accum) -> gates ->
// c_new (in place), h_new (ping-pong), optional copy of h_new to output.
// Block = 256 threads (8 waves). Block tile: 32 batch rows x 64 hidden cols
// x 4 gates. Each wave owns one 16x16 (m,n) tile with all 4 gate accums.
// Grid: (HS/64, BATCH/32) = (16, 4).
// ---------------------------------------------------------------------------
__global__ __launch_bounds__(256) void cell_kernel(
    const float* __restrict__ act, long act_stride,   // layer input rows
    const float* __restrict__ hprev,                  // recurrent h (read)
    float*       __restrict__ cst,                    // c state (in/out)
    float*       __restrict__ hout,                   // h write buffer
    const __bf16* __restrict__ wtA,                   // U^T bf16 [4096][1024]
    const __bf16* __restrict__ wtH,                   // V^T bf16 [4096][1024]
    const float* __restrict__ bias,                   // (4*HS,)
    const float* __restrict__ ubin,                   // (B,) binary u
    float*       __restrict__ out, long out_stride)   // optional h1 -> d_out
{
    __shared__ __bf16 atile[32 * LDS_STRIDE];

    const int tid  = threadIdx.x;
    const int lane = tid & 31;
    const int wv   = tid >> 5;
    const int m_base = blockIdx.y * 32;
    const int n_base = blockIdx.x * 64;
    const int mt = (wv & 1) * 16;                     // wave's m sub-tile
    const int nt = (wv >> 1) * 16;                    // wave's n sub-tile
    const int nl = n_base + nt + (lane & 15);         // this lane's hidden col

    v8f acc_i = {}, acc_f = {}, acc_g = {}, acc_o = {};

    // Staging: row = tid>>3 (0..31), col chunk = (tid&7)*4, step 32 cols.
    const int srow = tid >> 3;
    const int scol = (tid & 7) * 4;
    // A-fragment addressing (this code's contiguous k order, same as B):
    //   lanes 0-15 : K 0-15 ; lanes 16-31 : K 16-31
    const int arow = mt + (lane & 15);
    const int koff = (lane < 16) ? 0 : 16;
    // B-fragment addressing (contiguous k order, same as A):
    //   lanes 0-15 : K 0-15 ; lanes 16-31 : K 16-31
    const int kwo  = (lane < 16) ? 0 : 16;

    // 4 phases: {act, act, h, h} x K-chunks of 512.
    for (int ph = 0; ph < 4; ++ph) {
        const float*  src   = (ph < 2) ? act : hprev;
        const long    strd  = (ph < 2) ? act_stride : (long)HS;
        const __bf16* wt    = (ph < 2) ? wtA : wtH;
        const int     kbase = (ph & 1) * 512;

        __syncthreads();   // previous phase's LDS reads done
        {
            const float* sp = src + (long)(m_base + srow) * strd + kbase + scol;
            __bf16*      dp = &atile[srow * LDS_STRIDE + scol];
#pragma unroll
            for (int j = 0; j < 16; ++j) {
                const float4 v = *(const float4*)(sp + j * 32);
                __bf16* d = dp + j * 32;
                d[0] = (__bf16)v.x; d[1] = (__bf16)v.y;
                d[2] = (__bf16)v.z; d[3] = (__bf16)v.w;
            }
        }
        __syncthreads();   // tile visible to all waves

        // Branch-free, barrier-free WMMA sweep over this 512-wide K chunk.
        const __bf16* bp = wt + (size_t)nl * HS + kbase + kwo;
        const __bf16* ap = &atile[arow * LDS_STRIDE + koff];
#pragma unroll 4
        for (int kt = 0; kt < 16; ++kt) {
            const v8bf alo = *(const v8bf*)(ap);
            const v8bf ahi = *(const v8bf*)(ap + 8);
            const v16bf afrag = __builtin_shufflevector(
                alo, ahi, 0,1,2,3,4,5,6,7,8,9,10,11,12,13,14,15);

            // Gate g weight column lives at n + g*1024 -> +g*2^20 elements.
            const v16bf bfi = *(const v16bf*)(bp + (0u << 20));
            const v16bf bff = *(const v16bf*)(bp + (1u << 20));
            const v16bf bfg = *(const v16bf*)(bp + (2u << 20));
            const v16bf bfo = *(const v16bf*)(bp + (3u << 20));

            __builtin_prefetch(bp + 512, 0, 1);   // next K-chunk of this row

            acc_i = __builtin_amdgcn_wmma_f32_16x16x32_bf16(
                false, afrag, false, bfi, (short)0, acc_i, false, false);
            acc_f = __builtin_amdgcn_wmma_f32_16x16x32_bf16(
                false, afrag, false, bff, (short)0, acc_f, false, false);
            acc_g = __builtin_amdgcn_wmma_f32_16x16x32_bf16(
                false, afrag, false, bfg, (short)0, acc_g, false, false);
            acc_o = __builtin_amdgcn_wmma_f32_16x16x32_bf16(
                false, afrag, false, bfo, (short)0, acc_o, false, false);

            ap += 32;
            bp += 32;
        }
    }

    // Epilogue: gates + state update, all in fp32 registers.
    const float bi_b = bias[0 * HS + nl];
    const float bf_b = bias[1 * HS + nl];
    const float bg_b = bias[2 * HS + nl];
    const float bo_b = bias[3 * HS + nl];

#pragma unroll
    for (int r = 0; r < 8; ++r) {
        // C/D layout: lanes 0-15 -> M = r, lanes 16-31 -> M = r + 8; N = lane%16
        const int  m   = m_base + mt + ((lane < 16) ? r : r + 8);
        const long idx = (long)m * HS + nl;

        const float iv = sigmoidf_(acc_i[r] + bi_b);
        const float fv = sigmoidf_(acc_f[r] + bf_b);
        const float gv = tanhf(acc_g[r] + bg_b);
        const float ov = sigmoidf_(acc_o[r] + bo_b);

        const float cn = fv * cst[idx] + iv * gv;
        const float uu = ubin[m];
        const float hn = uu * (ov * tanhf(cn)) + (1.0f - uu) * hprev[idx];

        cst[idx]  = cn;
        hout[idx] = hn;
        if (out) out[(long)m * out_stride + nl] = hn;
    }
}

// ---------------------------------------------------------------------------
// Skip gate: cum_u = sigmoid(c1 @ skip_w + skip_b) per batch row, then
// u_t' = u*cum + (1-u)*(u_t + min(cum, 1-u_t)); u_bin' = round(u_t').
// One block per batch row.
// ---------------------------------------------------------------------------
__global__ __launch_bounds__(256) void skip_update(
    const float* __restrict__ c1, const float* __restrict__ skip_w,
    const float* __restrict__ skip_b, float* __restrict__ u_t,
    float* __restrict__ u_bin)
{
    __shared__ float red[256];
    const int b = blockIdx.x;
    float p = 0.0f;
    for (int k = threadIdx.x; k < HS; k += 256)
        p += c1[(long)b * HS + k] * skip_w[k];
    red[threadIdx.x] = p;
    __syncthreads();
    for (int s = 128; s > 0; s >>= 1) {
        if (threadIdx.x < s) red[threadIdx.x] += red[threadIdx.x + s];
        __syncthreads();
    }
    if (threadIdx.x == 0) {
        const float cum = sigmoidf_(red[0] + skip_b[0]);
        const float uo  = u_bin[b];
        const float ut  = u_t[b];
        const float utn = uo * cum + (1.0f - uo) * (ut + fminf(cum, 1.0f - ut));
        u_t[b]   = utn;
        u_bin[b] = rintf(utn);   // round-to-nearest-even, matches jnp.round
    }
}

// ---------------------------------------------------------------------------
extern "C" void kernel_launch(void* const* d_in, const int* in_sizes, int n_in,
                              void* d_out, int out_size, void* d_ws, size_t ws_size,
                              hipStream_t stream)
{
    const float* x  = (const float*)d_in[0];
    const float* U0 = (const float*)d_in[1];
    const float* V0 = (const float*)d_in[2];
    const float* b0 = (const float*)d_in[3];
    const float* U1 = (const float*)d_in[4];
    const float* V1 = (const float*)d_in[5];
    const float* b1 = (const float*)d_in[6];
    const float* sw = (const float*)d_in[7];
    const float* sb = (const float*)d_in[8];
    float* out = (float*)d_out;
    char*  ws  = (char*)d_ws;

    // Workspace layout (~37 MB):
    const size_t WTB = (size_t)4096 * 1024 * 2;       // 8 MB per bf16 W^T
    __bf16* wtU0 = (__bf16*)(ws + 0 * WTB);
    __bf16* wtV0 = (__bf16*)(ws + 1 * WTB);
    __bf16* wtU1 = (__bf16*)(ws + 2 * WTB);
    __bf16* wtV1 = (__bf16*)(ws + 3 * WTB);
    const size_t so = 4 * WTB;
    const size_t HB = (size_t)BATCH * HS * sizeof(float);   // 512 KB
    float* h0a = (float*)(ws + so + 0 * HB);
    float* h0b = (float*)(ws + so + 1 * HB);
    float* h1a = (float*)(ws + so + 2 * HB);
    float* h1b = (float*)(ws + so + 3 * HB);
    float* c0  = (float*)(ws + so + 4 * HB);
    float* c1  = (float*)(ws + so + 5 * HB);
    float* u_t = (float*)(ws + so + 6 * HB);
    float* u_b = (float*)(ws + so + 6 * HB + 512);

    // One-time per call: bf16 weight transpose + state init (deterministic).
    wt_transpose_bf16<<<16384, 256, 0, stream>>>(U0, wtU0);
    wt_transpose_bf16<<<16384, 256, 0, stream>>>(V0, wtV0);
    wt_transpose_bf16<<<16384, 256, 0, stream>>>(U1, wtU1);
    wt_transpose_bf16<<<16384, 256, 0, stream>>>(V1, wtV1);
    hipMemsetAsync(ws + so, 0, 6 * HB, stream);
    u_init<<<1, 128, 0, stream>>>(u_t, u_b);

    float* h0rw[2] = { h0a, h0b };
    float* h1rw[2] = { h1a, h1b };
    const dim3 grid(HS / 64, BATCH / 32);   // (16, 4)

    for (int t = 0; t < SEQ; ++t) {
        const int pr = t & 1, pw = pr ^ 1;
        // Layer 0: act = x[:, t, :]
        cell_kernel<<<grid, 256, 0, stream>>>(
            x + (long)t * HS, (long)SEQ * HS,
            h0rw[pr], c0, h0rw[pw],
            wtU0, wtV0, b0, u_b, nullptr, 0);
        // Layer 1: act = new h0; also writes h1 to out[:, t, :]
        cell_kernel<<<grid, 256, 0, stream>>>(
            h0rw[pw], (long)HS,
            h1rw[pr], c1, h1rw[pw],
            wtU1, wtV1, b1, u_b, out + (long)t * HS, (long)SEQ * HS);
        // Skip gate update (reads old u_b, writes new u_t / u_b for t+1)
        skip_update<<<BATCH, 256, 0, stream>>>(c1, sw, sb, u_t, u_b);
    }
}
